// MultiHeadSelfAttention_3289944949210
// MI455X (gfx1250) — matrix operand order repaired and checked by reading, not compiled, not run on hardware
//
#include <hip/hip_runtime.h>
#include <hip/hip_bf16.h>

typedef __attribute__((ext_vector_type(16))) __bf16       v16bf;
typedef __attribute__((ext_vector_type(8)))  float        v8f;
typedef __attribute__((ext_vector_type(4)))  unsigned int u32x4;
typedef __attribute__((ext_vector_type(4)))  float        f32x4;
typedef unsigned short ushort_t;

#define S_LEN 2048
#define H_DIM 1024
#define N_HEAD 16
#define HD 64
#define B_SZ 4
#define M_TOT (B_SZ * S_LEN)   /* 8192 */
#define NEGE -1e30f

// Fragment union: 16 bf16 = 8 VGPRs = 32 bytes
union Frag {
    v16bf    v;
    ushort_t u[16];
    u32x4    q[2];
};

union Pack8 {
    ushort_t u[8];
    u32x4    q;
};

__device__ inline ushort_t f32_to_bf16(float f) {
    union { float f; unsigned u; } x;
    x.f = f;
    unsigned r = x.u + 0x7fffu + ((x.u >> 16) & 1u);  // round to nearest even
    return (ushort_t)(r >> 16);
}

__device__ inline v8f wmma_bf16(const Frag& a, const Frag& b, v8f c) {
    return __builtin_amdgcn_wmma_f32_16x16x32_bf16(
        /*neg_a=*/false, a.v, /*neg_b=*/false, b.v,
        /*c_mod=*/(short)0, c, /*reuse_a=*/false, /*reuse_b=*/false);
}

// ---------------- fp32 -> bf16 conversion (vectorized x4) ----------------
__global__ __launch_bounds__(256) void cvt_bf16_kernel(const float* __restrict__ in,
                                                       ushort_t* __restrict__ out,
                                                       int n4) {
    int i = blockIdx.x * blockDim.x + threadIdx.x;
    if (i >= n4) return;
    f32x4 v = ((const f32x4*)in)[i];
    ushort_t* o = out + (size_t)i * 4;
    o[0] = f32_to_bf16(v.x); o[1] = f32_to_bf16(v.y);
    o[2] = f32_to_bf16(v.z); o[3] = f32_to_bf16(v.w);
}

// ---------------- bf16 GEMM:  out[M,N] = A[M,K] @ W[N,K]^T + bias ----------------
// WG tile 128(M) x 128(N); 8 waves in 2(m) x 4(n); wave tile 64 x 32 = 4x2 WMMA tiles.
// Output modes: outf != nullptr -> fp32 row-major
//               tmode == 1      -> bf16 transposed  out[col][row]  (V path)
//               else            -> bf16 row-major
__global__ __launch_bounds__(256) void gemm_bf16_kernel(const ushort_t* __restrict__ A,
                                                        const ushort_t* __restrict__ W,
                                                        const float* __restrict__ bias,
                                                        ushort_t* __restrict__ outb,
                                                        float* __restrict__ outf,
                                                        int M, int N, int K, int tmode) {
    const int tid  = threadIdx.x;
    const int wave = tid >> 5, lane = tid & 31;
    const int hl   = lane >> 4, lm = lane & 15;
    const int wm   = wave >> 2, wn = wave & 3;
    const int row0 = blockIdx.y * 128 + wm * 64;
    const int col0 = blockIdx.x * 128 + wn * 32;

    v8f acc[4][2] = {};

    for (int k0 = 0; k0 < K; k0 += 32) {
        Frag a[4], b[2];
#pragma unroll
        for (int im = 0; im < 4; ++im) {
            // A 16x32 frag: lane group hl: u[0..7] = K(hl*8..+7), u[8..15] = K(16+hl*8..+7)
            const ushort_t* p = A + (size_t)(row0 + im * 16 + lm) * K + k0 + hl * 8;
            a[im].q[0] = *(const u32x4*)p;
            a[im].q[1] = *(const u32x4*)(p + 16);
        }
#pragma unroll
        for (int in = 0; in < 2; ++in) {
            // B 32x16 frag: lane (hl,n): u[0..7] = K(hl*8..+7), u[8..15] = K(16+hl*8..+7)
            const ushort_t* p = W + (size_t)(col0 + in * 16 + lm) * K + k0 + hl * 8;
            b[in].q[0] = *(const u32x4*)p;
            b[in].q[1] = *(const u32x4*)(p + 16);
        }
        if (k0 + 32 < K) {
            __builtin_prefetch(A + (size_t)(row0 + lm) * K + k0 + 32, 0, 1);
            __builtin_prefetch(W + (size_t)(col0 + lm) * K + k0 + 32, 0, 1);
        }
#pragma unroll
        for (int im = 0; im < 4; ++im)
#pragma unroll
            for (int in = 0; in < 2; ++in)
                acc[im][in] = wmma_bf16(a[im], b[in], acc[im][in]);
    }

#pragma unroll
    for (int im = 0; im < 4; ++im)
#pragma unroll
        for (int in = 0; in < 2; ++in) {
            const int col = col0 + in * 16 + lm;
            const float bv = bias ? bias[col] : 0.f;
            if (outf) {                       // fp32 row-major (final projection)
#pragma unroll
                for (int r = 0; r < 8; ++r) {
                    const int row = row0 + im * 16 + r + 8 * hl;  // C-frag: M = r + 8*hl
                    outf[(size_t)row * N + col] = acc[im][in][r] + bv;
                }
            } else if (tmode == 1) {          // bf16 transposed: out[col][row], vectorized
                Pack8 pk;
#pragma unroll
                for (int r = 0; r < 8; ++r) pk.u[r] = f32_to_bf16(acc[im][in][r] + bv);
                const int rbase = row0 + im * 16 + 8 * hl;        // 8 consecutive rows
                *(u32x4*)(outb + (size_t)col * M + rbase) = pk.q;
            } else {                          // bf16 row-major
#pragma unroll
                for (int r = 0; r < 8; ++r) {
                    const int row = row0 + im * 16 + r + 8 * hl;
                    outb[(size_t)row * N + col] = f32_to_bf16(acc[im][in][r] + bv);
                }
            }
        }
}

// ---------------- flash attention over bf16 Q/K (row-major) and Vt (transposed) ----------------
// grid: (S/64, B*NH); 128 threads = 4 waves; wave owns 16 q-rows.
// Vt layout: Vt[col][m] with col = h*64+d (0..1023), m = b*S+s (0..8191).
__global__ __launch_bounds__(128) void attn_kernel(const ushort_t* __restrict__ Q,
                                                   const ushort_t* __restrict__ Kx,
                                                   const ushort_t* __restrict__ Vt,
                                                   const int* __restrict__ mask,
                                                   ushort_t* __restrict__ ao) {
    __shared__ ushort_t plds[4][16][32];  // per-wave P re-layout buffer (bf16)

    const int tid  = threadIdx.x;
    const int wave = tid >> 5, lane = tid & 31;
    const int hl   = lane >> 4, lm = lane & 15;
    const int bh = blockIdx.y;
    const int b  = bh >> 4, h = bh & 15;
    const int q0 = blockIdx.x * 64 + wave * 16;

    const ushort_t* Qh  = Q  + (size_t)b * S_LEN * H_DIM + h * HD;
    const ushort_t* Kh  = Kx + (size_t)b * S_LEN * H_DIM + h * HD;
    const ushort_t* VtH = Vt + (size_t)(h * HD) * M_TOT + (size_t)b * S_LEN;
    const int* mb = mask + b * S_LEN;

    // Q A-frags: 16 rows x 64 d  (two K=32 chunks)
    Frag aQ[2];
    {
        const ushort_t* p = Qh + (size_t)(q0 + lm) * H_DIM + hl * 8;
        aQ[0].q[0] = *(const u32x4*)p;
        aQ[0].q[1] = *(const u32x4*)(p + 16);
        aQ[1].q[0] = *(const u32x4*)(p + 32);
        aQ[1].q[1] = *(const u32x4*)(p + 48);
    }

    int rowIdx[8];
    int rowMi[8];
#pragma unroll
    for (int r = 0; r < 8; ++r) {
        rowIdx[r] = q0 + r + 8 * hl;
        rowMi[r]  = mb[rowIdx[r]];
    }

    float mstate[8], lstate[8];
    v8f accD[4] = {};
#pragma unroll
    for (int r = 0; r < 8; ++r) { mstate[r] = NEGE; lstate[r] = 0.f; }

    for (int j0 = 0; j0 < S_LEN; j0 += 32) {
        float et[2][8];
#pragma unroll
        for (int jt = 0; jt < 2; ++jt) {
            const int col = j0 + jt * 16 + lm;
            Frag bK0, bK1;
            const ushort_t* p = Kh + (size_t)col * H_DIM + hl * 8;
            bK0.q[0] = *(const u32x4*)p;        bK0.q[1] = *(const u32x4*)(p + 16);
            bK1.q[0] = *(const u32x4*)(p + 32); bK1.q[1] = *(const u32x4*)(p + 48);
            v8f sc = {};
            sc = wmma_bf16(aQ[0], bK0, sc);
            sc = wmma_bf16(aQ[1], bK1, sc);
            const int colm = mb[col];
#pragma unroll
            for (int r = 0; r < 8; ++r) {
                const bool valid = (rowMi[r] == 0) && (colm == 0) && (rowIdx[r] != col);
                et[jt][r] = valid ? sc[r] * 0.125f : NEGE;   // 1/sqrt(64)
            }
        }

        // online softmax, rows live in 16-lane groups (wave32: xor masks 1..8)
        float pt[2][8];
#pragma unroll
        for (int r = 0; r < 8; ++r) {
            float mx = fmaxf(et[0][r], et[1][r]);
#pragma unroll
            for (int s = 1; s < 16; s <<= 1) mx = fmaxf(mx, __shfl_xor(mx, s, 32));
            const float mnew  = fmaxf(mstate[r], mx);
            const float alpha = __expf(mstate[r] - mnew);
            const float p0 = (et[0][r] <= 0.5f * NEGE) ? 0.f : __expf(et[0][r] - mnew);
            const float p1 = (et[1][r] <= 0.5f * NEGE) ? 0.f : __expf(et[1][r] - mnew);
            float rs = p0 + p1;
#pragma unroll
            for (int s = 1; s < 16; s <<= 1) rs += __shfl_xor(rs, s, 32);
            lstate[r] = lstate[r] * alpha + rs;
            mstate[r] = mnew;
#pragma unroll
            for (int t = 0; t < 4; ++t) accD[t][r] *= alpha;
            pt[0][r] = p0; pt[1][r] = p1;
        }

        // C-frag -> A-frag re-layout of P through LDS (bf16)
#pragma unroll
        for (int jt = 0; jt < 2; ++jt)
#pragma unroll
            for (int r = 0; r < 8; ++r)
                plds[wave][r + 8 * hl][jt * 16 + lm] = f32_to_bf16(pt[jt][r]);
        asm volatile("s_wait_dscnt 0" ::: "memory");   // ds stores visible before ds loads

        Frag aP;
        {
            const ushort_t* p = &plds[wave][lm][hl * 8];
            aP.q[0] = *(const u32x4*)p;
            aP.q[1] = *(const u32x4*)(p + 16);
        }

        // accD[t] += P(16x32) x V(32x16): B-frag is contiguous in transposed Vt
#pragma unroll
        for (int t = 0; t < 4; ++t) {
            Frag bV;
            const ushort_t* p = VtH + (size_t)(t * 16 + lm) * M_TOT + j0 + hl * 8;
            bV.q[0] = *(const u32x4*)p;       // u[i<8] = V[k = j0+hl*8+i][d], u[i>=8] = V[k = j0+16+hl*8+i-8][d]
            bV.q[1] = *(const u32x4*)(p + 16);
            accD[t] = wmma_bf16(aP, bV, accD[t]);
        }
    }

    // epilogue: out = acc / l (0 if row fully masked); heads concat at cols h*64..
#pragma unroll
    for (int t = 0; t < 4; ++t)
#pragma unroll
        for (int r = 0; r < 8; ++r) {
            const float l = lstate[r];
            const float o = (l > 0.f) ? accD[t][r] / l : 0.f;
            ao[((size_t)b * S_LEN + rowIdx[r]) * H_DIM + h * HD + t * 16 + lm] = f32_to_bf16(o);
        }
}

// ---------------- launch ----------------
extern "C" void kernel_launch(void* const* d_in, const int* in_sizes, int n_in,
                              void* d_out, int out_size, void* d_ws, size_t ws_size,
                              hipStream_t stream) {
    const float* x    = (const float*)d_in[0];
    const int*   mask = (const int*)d_in[1];
    const float* Wq   = (const float*)d_in[2];
    const float* bq   = (const float*)d_in[3];
    const float* Wk   = (const float*)d_in[4];
    const float* bk   = (const float*)d_in[5];
    const float* Wv   = (const float*)d_in[6];
    const float* bv   = (const float*)d_in[7];
    const float* Wc   = (const float*)d_in[8];

    const int M = M_TOT;                      // 8192
    const size_t BSH = (size_t)M * H_DIM;     // 8,388,608
    const size_t HH  = (size_t)H_DIM * H_DIM;

    char* ws = (char*)d_ws;
    size_t off = 0;
    auto alloc = [&](size_t elems) {
        ushort_t* p = (ushort_t*)(ws + off);
        off += (elems * sizeof(ushort_t) + 255) & ~(size_t)255;
        return p;
    };
    ushort_t* xb  = alloc(BSH);
    ushort_t* wqb = alloc(HH);
    ushort_t* wkb = alloc(HH);
    ushort_t* wvb = alloc(HH);
    ushort_t* wcb = alloc(HH);
    ushort_t* Qb  = alloc(BSH);
    ushort_t* Kb  = alloc(BSH);
    ushort_t* Vt  = alloc(BSH);   // transposed: Vt[col][m]
    ushort_t* aob = alloc(BSH);

    // fp32 -> bf16
    {
        int n4 = (int)(BSH / 4);
        cvt_bf16_kernel<<<(n4 + 255) / 256, 256, 0, stream>>>(x, xb, n4);
        int w4 = (int)(HH / 4);
        cvt_bf16_kernel<<<(w4 + 255) / 256, 256, 0, stream>>>(Wq, wqb, w4);
        cvt_bf16_kernel<<<(w4 + 255) / 256, 256, 0, stream>>>(Wk, wkb, w4);
        cvt_bf16_kernel<<<(w4 + 255) / 256, 256, 0, stream>>>(Wv, wvb, w4);
        cvt_bf16_kernel<<<(w4 + 255) / 256, 256, 0, stream>>>(Wc, wcb, w4);
    }

    // projections: [8192,1024] = x @ W^T + b ; V written transposed
    dim3 ggrid(H_DIM / 128, M / 128);
    gemm_bf16_kernel<<<ggrid, 256, 0, stream>>>(xb, wqb, bq, Qb, nullptr, M, H_DIM, H_DIM, 0);
    gemm_bf16_kernel<<<ggrid, 256, 0, stream>>>(xb, wkb, bk, Kb, nullptr, M, H_DIM, H_DIM, 0);
    gemm_bf16_kernel<<<ggrid, 256, 0, stream>>>(xb, wvb, bv, Vt, nullptr, M, H_DIM, H_DIM, 1);

    // fused masked attention
    dim3 agrid(S_LEN / 64, B_SZ * N_HEAD);
    attn_kernel<<<agrid, 128, 0, stream>>>(Qb, Kb, Vt, mask, aob);

    // output projection -> fp32 d_out
    gemm_bf16_kernel<<<ggrid, 256, 0, stream>>>(aob, wcb, nullptr, nullptr, (float*)d_out,
                                                M, H_DIM, H_DIM, 0);
}
